// QConv_10565619548718
// MI455X (gfx1250) — matrix operand order repaired and checked by reading, not compiled, not run on hardware
//
#include <hip/hip_runtime.h>
#include <math.h>

#define PI_F 3.14159265358979f

// Problem constants (match reference setup_inputs)
#define BB 16
#define WW 64
#define HH 64
#define CC 8
#define DD 4
#define NX 63
#define NY 63
#define NPATCH (BB * NX * NY)      // 63504 = 3969 * 16 exactly
#define NTILES (NPATCH / 16)       // 3969
#define KFEAT 200                  // 8 channels * 25 bilinear features
#define KPAD 224                   // 7 chunks of 32
#define NCHUNK 7
#define BMAT_HALFS (KPAD * 16)     // 224 x 16 f16 weight panel (fragment order)
#define FSTRIDE 232                // feature row stride in halves: 464B = 29*16B, 29%16=13 -> conflict-free

typedef __attribute__((ext_vector_type(16))) _Float16 v16h;
typedef __attribute__((ext_vector_type(8)))  _Float16 v8h;
typedef __attribute__((ext_vector_type(8)))  float    v8f;

// ---------------- complex helpers ----------------
__device__ __forceinline__ float2 cmul(float2 a, float2 b) {
    return make_float2(a.x * b.x - a.y * b.y, a.x * b.y + a.y * b.x);
}
__device__ __forceinline__ float2 cadd(float2 a, float2 b) {
    return make_float2(a.x + b.x, a.y + b.y);
}

// CZPow(e_cz) then CXPow(e_cx) on (control_mask bh, target_mask bl)
__device__ void gate_pair(float2 amp[16], int bh, int bl, float ecz, float ecx) {
    float sph, cph;
    sincosf(PI_F * ecz, &sph, &cph);
    float2 ph = make_float2(cph, sph);               // e^{i*pi*e_cz}
    float th = 0.5f * PI_F * ecx;
    float st, ct;
    sincosf(th, &st, &ct);
    float2 cc = make_float2(ct * ct, st * ct);       // e^{i th} cos th
    float2 ss = make_float2(st * ct, -st * st);      // sin th e^{-i th}
    #pragma unroll
    for (int i = 0; i < 16; ++i)
        if ((i & bh) && (i & bl)) amp[i] = cmul(amp[i], ph);
    #pragma unroll
    for (int i = 0; i < 16; ++i) {
        if ((i & bh) && !(i & bl)) {
            int j = i | bl;
            float2 x = amp[i], y = amp[j];
            amp[i] = cadd(cmul(cc, x), cmul(ss, y));
            amp[j] = cadd(cmul(ss, x), cmul(cc, y));
        }
    }
}

// Exact reference 4-qubit simulation: returns <Z> on qubit 0.
// Qubit q <-> bit (3-q) of the flat index (qubit 0 = leading axis).
__device__ float simulate4(const float a[4], const float p[6]) {
    float c[4], s[4];
    #pragma unroll
    for (int q = 0; q < 4; ++q) sincosf(0.5f * PI_F * a[q], &s[q], &c[q]);
    float2 amp[16];
    #pragma unroll
    for (int i = 0; i < 16; ++i) {
        float2 v = make_float2(1.f, 0.f);
        #pragma unroll
        for (int q = 0; q < 4; ++q) {
            int bit = (i >> (3 - q)) & 1;
            float2 f = bit ? make_float2(0.f, -s[q]) : make_float2(c[q], 0.f);
            v = cmul(v, f);
        }
        amp[i] = v;
    }
    gate_pair(amp, 8, 4, p[0], p[1]);  // qubits (0,1)
    gate_pair(amp, 2, 1, p[2], p[3]);  // qubits (2,3)
    gate_pair(amp, 8, 2, p[4], p[5]);  // qubits (0,2)
    float z = 0.f;
    #pragma unroll
    for (int i = 0; i < 16; ++i) {
        float pr = amp[i].x * amp[i].x + amp[i].y * amp[i].y;
        z += (i & 8) ? -pr : pr;
    }
    return z;
}

// ---------------- setup: build weight panel in WMMA B-fragment order -------
// Logical panel: B[k][n], k = c*25 + mu*5 + nu (K=200 padded to 224), n = d.
// Fragment order: Bfrag[(chunk*32 + lane)*16 + h] where lane's halves hold
// B[k = chunk*32 + h + 16*(lane>=16)][n = lane%16]  (contiguous k order, matching the A loads).
// Each main-kernel lane then loads its v16h as 32 contiguous bytes.
__global__ __launch_bounds__(32) void qconv_setup(const float* __restrict__ kern,
                                                  _Float16* __restrict__ Bfrag) {
    const int t = threadIdx.x;        // 0..31 -> (d,c)
    for (int z = t; z < BMAT_HALFS; z += 32) Bfrag[z] = (_Float16)0.f;
    __syncthreads();

    const int d = t >> 3;
    const int c = t & 7;
    float p[6];
    #pragma unroll
    for (int m = 0; m < 6; ++m) p[m] = kern[(d * CC + c) * 6 + m];

    // basis pixel pairs -> feature vectors (1, C0, S0, S0C1, S0S1)
    const float A0[5] = {0.f, 1.f, 0.5f, 0.5f, 0.5f};
    const float A1[5] = {0.f, 0.f, 0.f,  1.f,  0.5f};
    float Z[5][5];
    for (int i = 0; i < 5; ++i)
        for (int j = 0; j < 5; ++j) {
            float a[4] = {A0[i], A1[i], A0[j], A1[j]};
            Z[i][j] = simulate4(a, p);
        }
    // F^{-1} for the basis above
    const float G[5][5] = {
        { 0.5f,  0.5f, 0.f,  0.f, 0.f},
        { 0.5f, -0.5f, 0.f,  0.f, 0.f},
        {-0.5f, -0.5f, 0.5f, 0.5f, 0.f},
        { 0.f,   0.f,  0.5f,-0.5f, 0.f},
        { 0.f,   0.f, -0.5f,-0.5f, 1.f}};
    float Qt[5][5];
    for (int m = 0; m < 5; ++m)
        for (int j = 0; j < 5; ++j) {
            float acc = 0.f;
            for (int i = 0; i < 5; ++i) acc += G[m][i] * Z[i][j];
            Qt[m][j] = acc;
        }
    for (int m = 0; m < 5; ++m)
        for (int n = 0; n < 5; ++n) {
            float acc = 0.f;
            for (int j = 0; j < 5; ++j) acc += Qt[m][j] * G[n][j];
            int kk   = c * 25 + m * 5 + n;       // logical K index
            int chnk = kk >> 5;
            int klo  = kk & 31;
            int lane = d + ((klo >= 16) ? 16 : 0);
            int h    = klo & 15;
            Bfrag[(chnk * 32 + lane) * 16 + h] = (_Float16)acc;
        }
}

// ---------------- main: feature build + WMMA GEMM + arccos epilogue --------
// One wave per block; each wave owns a 16-patch tile (M=16, K=224, N=16 with
// only N<4 valid). 7 x v_wmma_f32_16x16x32_f16, f32 accumulate.
__global__ __launch_bounds__(32) void qconv_main(const float* __restrict__ x,
                                                 const _Float16* __restrict__ Bfrag,
                                                 float* __restrict__ out) {
    __shared__ __align__(16) _Float16 feat[16][FSTRIDE];
    const int lane = threadIdx.x;
    const int n0 = blockIdx.x * 16;

    // ---- features: 16 patches x 8 channels = 128 items, 4 per lane ----
    #pragma unroll
    for (int r = 0; r < 4; ++r) {
        int item = lane + 32 * r;
        int pp = item >> 3;
        int ch = item & 7;
        int n = n0 + pp;
        int b = n / (NX * NY);
        int rem = n - b * (NX * NY);
        int i = rem / NY;
        int j = rem - i * NY;
        const float* px = x + ((b * WW + i) * HH + j) * CC + ch;
        float a0 = px[0];              // (di,dj)=(0,0)
        float a1 = px[CC];             // (0,1)
        float a2 = px[HH * CC];        // (1,0)
        float a3 = px[HH * CC + CC];   // (1,1)
        float s0, c0, s1, c1, s2, c2, s3, c3;
        sincosf(PI_F * a0, &s0, &c0);
        sincosf(PI_F * a1, &s1, &c1);
        sincosf(PI_F * a2, &s2, &c2);
        sincosf(PI_F * a3, &s3, &c3);
        float f01[5] = {1.f, c0, s0, s0 * c1, s0 * s1};
        float f23[5] = {1.f, c2, s2, s2 * c3, s2 * s3};
        int base = ch * 25;
        #pragma unroll
        for (int mu = 0; mu < 5; ++mu)
            #pragma unroll
            for (int nu = 0; nu < 5; ++nu)
                feat[pp][base + mu * 5 + nu] = (_Float16)(f01[mu] * f23[nu]);
    }
    // zero K padding columns 200..231: 16*32 = 512 = 16*32 entries
    #pragma unroll
    for (int z = 0; z < 16; ++z) {
        int idx = lane + z * 32;
        feat[idx >> 5][200 + (idx & 31)] = (_Float16)0.f;
    }
    __syncthreads();

    // ---- WMMA: acc[16x16] += A[16x32] * B[32x16], 7 K-chunks ----
    // A fragment (16-bit 16x32): lane holds row M=lane%16; its 16 halves are
// K = [kbase+16*hi, +8) and [kbase+16*hi+8, +8)  -> two b128 LDS loads.
    // B fragment: 32 contiguous bytes from the pre-swizzled panel.
    v8f acc = {};
    const int hi = (lane >= 16) ? 1 : 0;
    const int arow = lane & 15;          // A: M = lane % 16
    const int ncol = lane & 15;          // B/D: N = lane % 16
    const _Float16* rowp = &feat[arow][0];
    #pragma unroll
    for (int t = 0; t < NCHUNK; ++t) {
        const int kbase = t * 32;
        v8h alo = *(const v8h*)(rowp + kbase + 16 * hi);
        v8h ahi = *(const v8h*)(rowp + kbase + 16 * hi + 8);
        v16h Af = __builtin_shufflevector(alo, ahi, 0, 1, 2, 3, 4, 5, 6, 7,
                                          8, 9, 10, 11, 12, 13, 14, 15);
        v16h Bf = *(const v16h*)(Bfrag + (t * 32 + lane) * 16);
        acc = __builtin_amdgcn_wmma_f32_16x16x32_f16(
            /*neg_a=*/false, Af, /*neg_b=*/false, Bf,
            /*c_mod=*/(short)0, acc, /*reuse_a=*/false, /*reuse_b=*/false);
    }

    // ---- epilogue: lanes holding valid depth columns (N<4) store results ----
    if (ncol < DD) {
        const int mbase = hi ? 8 : 0;    // C/D: M = r + 8*(lane>=16)
        #pragma unroll
        for (int r = 0; r < 8; ++r) {
            float v = acc[r];
            v = fminf(fmaxf(v, -1.0f + 1e-5f), 1.0f - 1e-5f);
            out[(n0 + mbase + r) * DD + ncol] = acosf(v) * (1.0f / PI_F);
        }
    }
}

extern "C" void kernel_launch(void* const* d_in, const int* in_sizes, int n_in,
                              void* d_out, int out_size, void* d_ws, size_t ws_size,
                              hipStream_t stream) {
    const float* x    = (const float*)d_in[0];   // [16,64,64,8] f32
    const float* kern = (const float*)d_in[1];   // [4,8,6] f32
    float* out = (float*)d_out;                  // [16,63,63,4] f32
    _Float16* Bfrag = (_Float16*)d_ws;           // 224x16 f16 panel, fragment order (7168 B)

    qconv_setup<<<1, 32, 0, stream>>>(kern, Bfrag);
    qconv_main<<<NTILES, 32, 0, stream>>>(x, Bfrag, out);
}
